// GAT_46024869544126
// MI455X (gfx1250) — matrix operand order repaired and checked by reading, not compiled, not run on hardware
//
#include <hip/hip_runtime.h>
#include <hip/hip_bf16.h>

#define NN 50000        // nodes
#define NE 800000       // raw edges
#define ET (NE + NN)    // edges + self loops = 850000
#define IN_DIM 512
#define FD 64           // feature size per layer
#define NG 128          // graphs
#define OD 10           // out dim
#define SLOPE 0.2f

typedef __attribute__((ext_vector_type(16))) _Float16 v16h;
typedef __attribute__((ext_vector_type(8)))  float    v8f;

union Frag16 { v16h v; uint4 q[2]; };
union H2U    { _Float16 h[2]; unsigned int u; };

// ------------- conversion: f32 -> f16, float4 / 8B-packed stores --------------
__global__ void k_f32_to_f16(const float* __restrict__ in, _Float16* __restrict__ out, int n4) {
  int i = blockIdx.x * blockDim.x + threadIdx.x;
  if (i >= n4) return;
  float4 v = ((const float4*)in)[i];
  H2U p0, p1;
  p0.h[0] = (_Float16)v.x; p0.h[1] = (_Float16)v.y;
  p1.h[0] = (_Float16)v.z; p1.h[1] = (_Float16)v.w;
  ((uint2*)out)[i] = make_uint2(p0.u, p1.u);
}

// ------------- pre-swizzle weight W[K][64] into B-fragment layout -------------
// Fragment (kt, nt): lane l holds column n = nt*16 + (l&15), K pairs
// K = kt*32 + (l>>4)*8 + (j>>2)*16 + 2*(j&3) (lo) / +1 (hi) for uint slot j in 0..7.
// Stored lane-major so the GEMM wave reads 32 contiguous bytes per fragment.
__global__ void k_swizzle_w(const float* __restrict__ W, unsigned int* __restrict__ Bsw, int K) {
  int idx = blockIdx.x * blockDim.x + threadIdx.x;
  int total = (K >> 5) * 4 * 32 * 8;
  if (idx >= total) return;
  int j    = idx & 7;
  int lane = (idx >> 3) & 31;
  int nt   = (idx >> 8) & 3;
  int kt   = idx >> 10;
  int n = nt * 16 + (lane & 15);
  int k = kt * 32 + (lane >> 4) * 8 + (j >> 2) * 16 + 2 * (j & 3);
  H2U p;
  p.h[0] = (_Float16)W[(size_t)k * FD + n];
  p.h[1] = (_Float16)W[(size_t)(k + 1) * FD + n];
  Bsw[idx] = p.u;
}

// ---------------- GEMM: C[M x 64] = A[M x K] @ W, via WMMA f16 ----------------
// One wave per 16-row tile; 4 accumulators cover N = 64.
__global__ __launch_bounds__(256) void k_gemm_wmma(
    const _Float16* __restrict__ A, const unsigned int* __restrict__ Bsw,
    float* __restrict__ C, int M, int K) {
  const int lane  = threadIdx.x & 31;
  const int wave  = threadIdx.x >> 5;
  const int tileM = blockIdx.x * 8 + wave;
  if (tileM * 16 >= M) return;                 // wave-uniform exit
  const int half16 = lane >> 4;                // 0 or 1
  const int mloc   = lane & 15;
  const int row    = tileM * 16 + mloc;        // A row this lane feeds
  const int KT     = K >> 5;

  v8f c0 = {}, c1 = {}, c2 = {}, c3 = {};
  const uint4* Bq = (const uint4*)Bsw;

  for (int kt = 0; kt < KT; ++kt) {
    Frag16 a;
    const uint4* Ap = (const uint4*)(A + (size_t)row * K + kt * 32 + half16 * 8);
    a.q[0] = Ap[0];        // K offsets +0..7   (16 B)
    a.q[1] = Ap[2];        // K offsets +16..23 (16 B)

    Frag16 b0, b1, b2, b3;
    const uint4* Bp = Bq + ((size_t)(kt * 4) * 32 + lane) * 2;
    b0.q[0] = Bp[0];      b0.q[1] = Bp[1];
    b1.q[0] = Bp[64];     b1.q[1] = Bp[65];    // next nt: +32 lanes * 2 uint4
    b2.q[0] = Bp[128];    b2.q[1] = Bp[129];
    b3.q[0] = Bp[192];    b3.q[1] = Bp[193];

    c0 = __builtin_amdgcn_wmma_f32_16x16x32_f16(false, a.v, false, b0.v, (short)0, c0, false, false);
    c1 = __builtin_amdgcn_wmma_f32_16x16x32_f16(false, a.v, false, b1.v, (short)0, c1, false, false);
    c2 = __builtin_amdgcn_wmma_f32_16x16x32_f16(false, a.v, false, b2.v, (short)0, c2, false, false);
    c3 = __builtin_amdgcn_wmma_f32_16x16x32_f16(false, a.v, false, b3.v, (short)0, c3, false, false);
  }

  // D layout: lanes 0-15 -> M rows 0..7, lanes 16-31 -> rows 8..15; col = lane&15
  const int rowb = tileM * 16 + half16 * 8;
  float* Cr = C + (size_t)rowb * FD + mloc;
#pragma unroll
  for (int r = 0; r < 8; ++r) {
    Cr[r * FD +  0] = c0[r];
    Cr[r * FD + 16] = c1[r];
    Cr[r * FD + 32] = c2[r];
    Cr[r * FD + 48] = c3[r];
  }
}

// ---------------- per-node attention projections alpha_s/alpha_d --------------
__global__ void k_alphas(const float* __restrict__ h, const float* __restrict__ a_s,
                         const float* __restrict__ a_d, float* __restrict__ as_,
                         float* __restrict__ ad_, int n) {
  int i = blockIdx.x * blockDim.x + threadIdx.x;
  if (i >= n) return;
  const float4* hr = (const float4*)(h + (size_t)i * FD);
  const float4* s4 = (const float4*)a_s;
  const float4* d4 = (const float4*)a_d;
  float s = 0.f, d = 0.f;
#pragma unroll
  for (int f = 0; f < FD / 4; ++f) {
    float4 v = hr[f], sa = s4[f], da = d4[f];
    s += v.x * sa.x + v.y * sa.y + v.z * sa.z + v.w * sa.w;
    d += v.x * da.x + v.y * da.y + v.z * da.z + v.w * da.w;
  }
  as_[i] = s; ad_[i] = d;
}

// ------------- per-layer init: zero out-buffer, m=-inf, denom=0 ---------------
__global__ void k_layer_init(float* __restrict__ xl, float* __restrict__ m,
                             float* __restrict__ den, int n) {
  int i = blockIdx.x * blockDim.x + threadIdx.x;
  if (i < n * (FD / 4)) ((float4*)xl)[i] = make_float4(0.f, 0.f, 0.f, 0.f);
  if (i < n) { m[i] = __int_as_float(0xFF800000); den[i] = 0.f; }
}

__device__ __forceinline__ void edge_nodes(const long long* __restrict__ ei, int e,
                                           int& s, int& d) {
  if (e < NE) { s = (int)ei[e]; d = (int)ei[NE + e]; }
  else        { s = d = e - NE; }   // implicit self loops
}

__device__ __forceinline__ void atomicMaxF(float* addr, float v) {
  if (v >= 0.f) atomicMax((int*)addr, __float_as_int(v));
  else          atomicMin((unsigned int*)addr, __float_as_uint(v));
}

// ----- pass 1: e = leaky_relu(as[src]+ad[dst]); segment max by dst -----------
__global__ void k_edge_logits(const long long* __restrict__ ei, const float* __restrict__ as_,
                              const float* __restrict__ ad_, float* __restrict__ ebuf,
                              float* __restrict__ mbuf) {
  int e = blockIdx.x * blockDim.x + threadIdx.x;
  if (e >= ET) return;
  int s, d; edge_nodes(ei, e, s, d);
  float v = as_[s] + ad_[d];
  v = v > 0.f ? v : SLOPE * v;
  ebuf[e] = v;
  atomicMaxF(mbuf + d, v);
}

// ----- pass 2: ex = exp(e - m[dst]); segment sum of ex by dst ----------------
__global__ void k_edge_exp(const long long* __restrict__ ei, float* __restrict__ ebuf,
                           const float* __restrict__ mbuf, float* __restrict__ dbuf) {
  int e = blockIdx.x * blockDim.x + threadIdx.x;
  if (e >= ET) return;
  int s, d; edge_nodes(ei, e, s, d);
  float ex = __expf(ebuf[e] - mbuf[d]);
  ebuf[e] = ex;
  atomicAdd(dbuf + d, ex);
}

// ----- pass 3: out[dst] += h[src] * ex/(den[dst]+eps) -------------------------
// Thread per (edge, 4-feature chunk): one GLOBAL_LOAD_B128 + 4 L2 atomics.
__global__ void k_edge_agg(const long long* __restrict__ ei, const float* __restrict__ ebuf,
                           const float* __restrict__ dbuf, const float* __restrict__ h,
                           float* __restrict__ xl) {
  int idx = blockIdx.x * blockDim.x + threadIdx.x;
  int e = idx >> 4, f4 = idx & 15;
  if (e >= ET) return;
  int s, d; edge_nodes(ei, e, s, d);
  float coef = ebuf[e] / (dbuf[d] + 1e-16f);
  float4 hv = ((const float4*)(h + (size_t)s * FD))[f4];
  float* dst = xl + (size_t)d * FD + f4 * 4;
  atomicAdd(dst + 0, hv.x * coef);
  atomicAdd(dst + 1, hv.y * coef);
  atomicAdd(dst + 2, hv.z * coef);
  atomicAdd(dst + 3, hv.w * coef);
}

// ----- bias + relu (float4), and emit f16 copy for next layer's GEMM ----------
__global__ void k_bias_relu_cvt(float* __restrict__ xl, const float* __restrict__ b,
                                _Float16* __restrict__ h16, int n) {
  int i = blockIdx.x * blockDim.x + threadIdx.x;
  if (i >= n * (FD / 4)) return;
  float4 v  = ((float4*)xl)[i];
  float4 bb = ((const float4*)b)[i & (FD / 4 - 1)];
  v.x = fmaxf(v.x + bb.x, 0.f);
  v.y = fmaxf(v.y + bb.y, 0.f);
  v.z = fmaxf(v.z + bb.z, 0.f);
  v.w = fmaxf(v.w + bb.w, 0.f);
  ((float4*)xl)[i] = v;
  H2U p0, p1;
  p0.h[0] = (_Float16)v.x; p0.h[1] = (_Float16)v.y;
  p1.h[0] = (_Float16)v.z; p1.h[1] = (_Float16)v.w;
  ((uint2*)h16)[i] = make_uint2(p0.u, p1.u);
}

__global__ void k_zero(float* __restrict__ p, int n) {
  int i = blockIdx.x * blockDim.x + threadIdx.x;
  if (i < n) p[i] = 0.f;
}

__global__ void k_counts(const long long* __restrict__ batch, float* __restrict__ cnt) {
  int i = blockIdx.x * blockDim.x + threadIdx.x;
  if (i < NN) atomicAdd(cnt + (int)batch[i], 1.0f);
}

// thread per (node, 4-feature chunk of the 192-wide concat)
__global__ void k_pool(const long long* __restrict__ batch, const float* __restrict__ x1,
                       const float* __restrict__ x2, const float* __restrict__ x3,
                       float* __restrict__ pooled) {
  int idx = blockIdx.x * blockDim.x + threadIdx.x;
  if (idx >= NN * 48) return;
  int i = idx / 48, f4 = idx % 48;
  const float* xs = (f4 < 16) ? x1 : (f4 < 32) ? x2 : x3;
  float4 v = ((const float4*)(xs + (size_t)i * FD))[f4 & 15];
  float* dst = pooled + (size_t)(int)batch[i] * 192 + f4 * 4;
  atomicAdd(dst + 0, v.x);
  atomicAdd(dst + 1, v.y);
  atomicAdd(dst + 2, v.z);
  atomicAdd(dst + 3, v.w);
}

__global__ void k_head(const float* __restrict__ pooled, const float* __restrict__ cnt,
                       const float* __restrict__ lw, const float* __restrict__ lb,
                       float* __restrict__ out) {
  int g = blockIdx.x * blockDim.x + threadIdx.x;
  if (g >= NG) return;
  float inv = 1.0f / fmaxf(cnt[g], 1.0f);
  float logit[OD];
#pragma unroll
  for (int j = 0; j < OD; ++j) logit[j] = lb[j];
  const float* pg = pooled + (size_t)g * 192;
  for (int f = 0; f < 192; ++f) {
    float v = pg[f] * inv;
#pragma unroll
    for (int j = 0; j < OD; ++j) logit[j] += v * lw[f * OD + j];
  }
  float mx = logit[0];
#pragma unroll
  for (int j = 1; j < OD; ++j) mx = fmaxf(mx, logit[j]);
  float s = 0.f;
#pragma unroll
  for (int j = 0; j < OD; ++j) { logit[j] = __expf(logit[j] - mx); s += logit[j]; }
#pragma unroll
  for (int j = 0; j < OD; ++j) out[(size_t)g * OD + j] = logit[j] / s;
}

// -----------------------------------------------------------------------------
static inline int cdiv(long long a, long long b) { return (int)((a + b - 1) / b); }

extern "C" void kernel_launch(void* const* d_in, const int* in_sizes, int n_in,
                              void* d_out, int out_size, void* d_ws, size_t ws_size,
                              hipStream_t stream) {
  (void)in_sizes; (void)n_in; (void)out_size; (void)ws_size;

  const float*     feat = (const float*)d_in[0];
  const long long* ei   = (const long long*)d_in[1];
  const long long* bat  = (const long long*)d_in[2];
  const float* W[3]  = { (const float*)d_in[3],  (const float*)d_in[7],  (const float*)d_in[11] };
  const float* AS[3] = { (const float*)d_in[4],  (const float*)d_in[8],  (const float*)d_in[12] };
  const float* AD[3] = { (const float*)d_in[5],  (const float*)d_in[9],  (const float*)d_in[13] };
  const float* BB[3] = { (const float*)d_in[6],  (const float*)d_in[10], (const float*)d_in[14] };
  const float* lin_w = (const float*)d_in[15];
  const float* lin_b = (const float*)d_in[16];
  float* out = (float*)d_out;

  // workspace carve-out
  char* ws = (char*)d_ws;
  size_t off = 0;
  auto carve = [&](size_t bytes) -> void* {
    void* p = ws + off; off += (bytes + 255) & ~(size_t)255; return p;
  };
  _Float16* feat16 = (_Float16*)carve((size_t)NN * IN_DIM * 2);
  _Float16* h16    = (_Float16*)carve((size_t)NN * FD * 2);
  float* hbuf  = (float*)carve((size_t)NN * FD * 4);
  float* x1    = (float*)carve((size_t)NN * FD * 4);
  float* x2    = (float*)carve((size_t)NN * FD * 4);
  float* x3    = (float*)carve((size_t)NN * FD * 4);
  float* as_   = (float*)carve((size_t)NN * 4);
  float* ad_   = (float*)carve((size_t)NN * 4);
  float* mbuf  = (float*)carve((size_t)NN * 4);
  float* dbuf  = (float*)carve((size_t)NN * 4);
  float* ebuf  = (float*)carve((size_t)ET * 4);
  unsigned int* wsw1 = (unsigned int*)carve((size_t)(IN_DIM / 32) * 4 * 32 * 8 * 4);
  unsigned int* wsw2 = (unsigned int*)carve((size_t)(FD / 32) * 4 * 32 * 8 * 4);
  unsigned int* wsw3 = (unsigned int*)carve((size_t)(FD / 32) * 4 * 32 * 8 * 4);
  float* pooled = (float*)carve((size_t)NG * 192 * 4);
  float* counts = (float*)carve((size_t)NG * 4);

  const int B = 256;

  // prep: fp16 features + swizzled weights
  k_f32_to_f16<<<cdiv((long long)NN * IN_DIM / 4, B), B, 0, stream>>>(feat, feat16, NN * IN_DIM / 4);
  k_swizzle_w<<<cdiv((IN_DIM / 32) * 4 * 32 * 8, B), B, 0, stream>>>(W[0], wsw1, IN_DIM);
  k_swizzle_w<<<cdiv((FD / 32) * 4 * 32 * 8, B), B, 0, stream>>>(W[1], wsw2, FD);
  k_swizzle_w<<<cdiv((FD / 32) * 4 * 32 * 8, B), B, 0, stream>>>(W[2], wsw3, FD);

  float* XL[3] = { x1, x2, x3 };
  const unsigned int* WS[3] = { wsw1, wsw2, wsw3 };
  const int KS[3] = { IN_DIM, FD, FD };

  for (int l = 0; l < 3; ++l) {
    const _Float16* Ain = (l == 0) ? feat16 : h16;
    // h = x @ W  (WMMA)
    k_gemm_wmma<<<cdiv(NN / 16, 8), B, 0, stream>>>(Ain, WS[l], hbuf, NN, KS[l]);
    // per-node attention scalars
    k_alphas<<<cdiv(NN, B), B, 0, stream>>>(hbuf, AS[l], AD[l], as_, ad_, NN);
    // init segment buffers + output
    k_layer_init<<<cdiv((long long)NN * (FD / 4), B), B, 0, stream>>>(XL[l], mbuf, dbuf, NN);
    // segment softmax + weighted aggregation
    k_edge_logits<<<cdiv(ET, B), B, 0, stream>>>(ei, as_, ad_, ebuf, mbuf);
    k_edge_exp<<<cdiv(ET, B), B, 0, stream>>>(ei, ebuf, mbuf, dbuf);
    k_edge_agg<<<cdiv((long long)ET * 16, B), B, 0, stream>>>(ei, ebuf, dbuf, hbuf, XL[l]);
    // bias + relu, emit f16 for next layer
    k_bias_relu_cvt<<<cdiv((long long)NN * (FD / 4), B), B, 0, stream>>>(XL[l], BB[l], h16, NN);
  }

  // mean pool + classifier head
  k_zero<<<cdiv(NG * 192, B), B, 0, stream>>>(pooled, NG * 192);
  k_zero<<<cdiv(NG, B), B, 0, stream>>>(counts, NG);
  k_counts<<<cdiv(NN, B), B, 0, stream>>>(bat, counts);
  k_pool<<<cdiv((long long)NN * 48, B), B, 0, stream>>>(bat, x1, x2, x3, pooled);
  k_head<<<cdiv(NG, B), B, 0, stream>>>(pooled, counts, lin_w, lin_b, out);
}
